// DenseLayerWithComplexNeurons_45749991637479
// MI455X (gfx1250) — matrix operand order repaired and checked by reading, not compiled, not run on hardware
//
#include <hip/hip_runtime.h>
#include <stdint.h>

typedef __attribute__((ext_vector_type(16))) __bf16 v16bf;
typedef __attribute__((ext_vector_type(8)))  float  v8f;
typedef unsigned short u16;
typedef unsigned int   u32;
typedef __attribute__((address_space(3))) u16 lds_u16;

#define M_TOT 8192   // B*S
#define N_TOT 4096   // A*DOUT
#define K_TOT 1024   // DIN
#define DOUT  1024
#define BM 128
#define BN 128
#define BK 32
#define SA 40                  // padded LDS row stride in elements (conflict-free)
#define KTILES (K_TOT / BK)    // 32

union FragU { uint4 u[2]; v16bf v; };

// ---------------- fp32 -> bf16 (round-to-nearest-even), vectorized ----------------
__global__ __launch_bounds__(256)
void f32_to_bf16_kernel(const float* __restrict__ src, u16* __restrict__ dst, int n4) {
  int i = blockIdx.x * blockDim.x + threadIdx.x;
  const int stride = gridDim.x * blockDim.x;
  const uint4* s = (const uint4*)src;
  uint2* d = (uint2*)dst;
  for (; i < n4; i += stride) {
    uint4 v = s[i];
    u32 r[4] = {v.x, v.y, v.z, v.w};
    u16 o[4];
#pragma unroll
    for (int j = 0; j < 4; ++j) {
      u32 u = r[j];
      u = u + 0x7FFFu + ((u >> 16) & 1u);   // RNE
      o[j] = (u16)(u >> 16);
    }
    uint2 w;
    w.x = (u32)o[0] | ((u32)o[1] << 16);
    w.y = (u32)o[2] | ((u32)o[3] << 16);
    d[i] = w;
  }
}

// ---------------- fused bf16 WMMA GEMM + per-cell-type MLP epilogue ----------------
__global__ __launch_bounds__(256)
void fused_gemm_mlp_kernel(const u16* __restrict__ xb,   // (8192,1024) bf16
                           const u16* __restrict__ wb,   // (4096,1024) bf16
                           const float* __restrict__ bias,
                           const float* __restrict__ cw1, // (T,A,H) = (4,4,8)
                           const float* __restrict__ cb1, // (4,8)
                           const float* __restrict__ cw2, // (4,8)
                           const float* __restrict__ cb2, // (4,)
                           float* __restrict__ out) {     // (8192,1024)
  __shared__ uint4 smem[4224];                 // 67,584 bytes
  u16*  As = (u16*)smem;                       // [2][BM*SA]  (20,480 B)
  u16*  Bs = As + 2 * BM * SA;                 // [2][BN*SA]  (20,480 B)
  float* zs = (float*)smem;                    // [128][132]  aliased after K loop

  const int tid  = threadIdx.x;
  const int lane = tid & 31;
  const int wave = tid >> 5;
  const int wm   = wave & 3;     // 4 row-groups of 32
  const int wn   = wave >> 2;    // 2 col-groups of 64
  const int m0   = blockIdx.y * BM;
  const int n0   = blockIdx.x * BN;

  // global->LDS async staging: each thread moves 32B of A and 32B of B per tile
  const int grow = tid >> 1;             // 0..127
  const int gkc  = (tid & 1) * 16;       // 0 or 16

  // async copy: INST_OFFSET is added to BOTH the LDS and global addresses,
  // so one (lds,global) address pair covers both 16B chunks.
  auto issue_tile = [&](int kt, int b) {
    const u16* ga = xb + (size_t)(m0 + grow) * K_TOT + kt * BK + gkc;
    u32 la = (u32)(size_t)(lds_u16*)(As + b * BM * SA + grow * SA + gkc);
    asm volatile("global_load_async_to_lds_b128 %0, %1, off"
                 :: "v"(la), "v"(ga) : "memory");
    asm volatile("global_load_async_to_lds_b128 %0, %1, off offset:16"
                 :: "v"(la), "v"(ga) : "memory");
    const u16* gb = wb + (size_t)(n0 + grow) * K_TOT + kt * BK + gkc;
    u32 lb = (u32)(size_t)(lds_u16*)(Bs + b * BN * SA + grow * SA + gkc);
    asm volatile("global_load_async_to_lds_b128 %0, %1, off"
                 :: "v"(lb), "v"(gb) : "memory");
    asm volatile("global_load_async_to_lds_b128 %0, %1, off offset:16"
                 :: "v"(lb), "v"(gb) : "memory");
  };

  v8f acc[2][4];
  const v8f vzero = {0.f, 0.f, 0.f, 0.f, 0.f, 0.f, 0.f, 0.f};
#pragma unroll
  for (int mi = 0; mi < 2; ++mi)
#pragma unroll
    for (int ni = 0; ni < 4; ++ni) acc[mi][ni] = vzero;

  issue_tile(0, 0);

  int buf = 0;
#pragma unroll 1
  for (int kt = 0; kt < KTILES; ++kt) {
    asm volatile("s_wait_asynccnt 0x0" ::: "memory");  // this wave's tile kt landed
    __syncthreads();                                   // all waves' tile kt visible;
                                                       // all waves done reading buf^1
    if (kt + 1 < KTILES) issue_tile(kt + 1, buf ^ 1);  // fill next buffer during WMMAs

    // A fragment: lanes 0-15 -> K {0..7,16..23}; lanes 16-31 -> K {8..15,24..31}
    // B fragment: lanes 0-15 -> K {0..7,16..23}; lanes 16-31 -> K {8..15,24..31} (lane%16 = column)
    const int halfA = (lane & 16) ? 8 : 0;
    const int halfB = (lane & 16) ? 8 : 0;
    FragU a[2], b[4];
#pragma unroll
    for (int mi = 0; mi < 2; ++mi) {
      const u16* base = As + buf * BM * SA + (wm * 32 + mi * 16 + (lane & 15)) * SA + halfA;
      a[mi].u[0] = *(const uint4*)(base);
      a[mi].u[1] = *(const uint4*)(base + 16);
    }
#pragma unroll
    for (int ni = 0; ni < 4; ++ni) {
      const u16* base = Bs + buf * BN * SA + (wn * 64 + ni * 16 + (lane & 15)) * SA + halfB;
      b[ni].u[0] = *(const uint4*)(base);
      b[ni].u[1] = *(const uint4*)(base + 16);
    }

#pragma unroll
    for (int mi = 0; mi < 2; ++mi)
#pragma unroll
      for (int ni = 0; ni < 4; ++ni)
        acc[mi][ni] = __builtin_amdgcn_wmma_f32_16x16x32_bf16(
            false, a[mi].v, false, b[ni].v, (short)0, acc[mi][ni], false, false);

    buf ^= 1;
  }
  __syncthreads();               // done reading tile LDS; safe to alias as zs

  // C layout: VGPR r -> M = r + 8*(lane/16), N = lane%16
#pragma unroll
  for (int mi = 0; mi < 2; ++mi)
#pragma unroll
    for (int ni = 0; ni < 4; ++ni) {
      const int col   = wn * 64 + ni * 16 + (lane & 15);
      const int rbase = wm * 32 + mi * 16 + ((lane & 16) ? 8 : 0);
#pragma unroll
      for (int r = 0; r < 8; ++r)
        zs[(rbase + r) * 132 + col] = acc[mi][ni][r];
    }
  __syncthreads();

  // ---------- per-cell-type MLP epilogue (t constant per N tile) ----------
  const int t   = n0 >> 10;      // neuron block of 1024 z-cols per cell type
  const int nb0 = n0 >> 2;       // first output neuron of this tile
  float w1[4][8], b1[8], w2[8];
#pragma unroll
  for (int aa = 0; aa < 4; ++aa)
#pragma unroll
    for (int h = 0; h < 8; ++h) w1[aa][h] = cw1[t * 32 + aa * 8 + h];
#pragma unroll
  for (int h = 0; h < 8; ++h) { b1[h] = cb1[t * 8 + h]; w2[h] = cw2[t * 8 + h]; }
  const float b2v = cb2[t];

#pragma unroll 1
  for (int i = 0; i < 16; ++i) {
    const int row = wave * 16 + i;   // 8 waves * 16 = 128 rows
    const int col = lane;            // 32 output neurons per tile
    float4 zv = *(const float4*)(zs + row * 132 + col * 4);
    float4 bv = *(const float4*)(bias + n0 + col * 4);
    const float z0 = zv.x + bv.x, z1 = zv.y + bv.y, z2 = zv.z + bv.z, z3 = zv.w + bv.w;
    float o = b2v;
#pragma unroll
    for (int h = 0; h < 8; ++h) {
      float pre = z0 * w1[0][h] + z1 * w1[1][h] + z2 * w1[2][h] + z3 * w1[3][h] + b1[h];
      o += tanhf(pre) * w2[h];
    }
    out[(size_t)(m0 + row) * DOUT + nb0 + col] = o;   // 128B coalesced per wave
  }
}

extern "C" void kernel_launch(void* const* d_in, const int* in_sizes, int n_in,
                              void* d_out, int out_size, void* d_ws, size_t ws_size,
                              hipStream_t stream) {
  const float* x      = (const float*)d_in[0];
  const float* weight = (const float*)d_in[1];
  const float* bias   = (const float*)d_in[2];
  const float* cw1    = (const float*)d_in[3];
  const float* cb1    = (const float*)d_in[4];
  const float* cw2    = (const float*)d_in[5];
  const float* cb2    = (const float*)d_in[6];
  float* out = (float*)d_out;

  // workspace: bf16 copies of x (16 MB) and weight (8 MB)
  u16* xb = (u16*)d_ws;
  u16* wb = xb + (size_t)M_TOT * K_TOT;

  f32_to_bf16_kernel<<<4096, 256, 0, stream>>>(x, xb, M_TOT * K_TOT / 4);
  f32_to_bf16_kernel<<<2048, 256, 0, stream>>>(weight, wb, N_TOT * K_TOT / 4);

  dim3 grid(N_TOT / BN, M_TOT / BM);   // 32 x 64
  fused_gemm_mlp_kernel<<<grid, 256, 0, stream>>>(xb, wb, bias, cw1, cb1, cw2, cb2, out);
}
